// GraspabilityGuidedTokenizer_49211735277866
// MI455X (gfx1250) — compile-verified, matrix operand order checked by reading, not run on hardware
//
#include <hip/hip_runtime.h>
#include <hip/hip_bf16.h>

// ---------------------------------------------------------------------------
// GraspabilityGuidedTokenizer for MI455X (gfx1250, wave32)
//   B=8, K=65536, C=256, H=128, T=2048, D=512
// ---------------------------------------------------------------------------

typedef __attribute__((ext_vector_type(16))) __bf16        v16bf;
typedef __attribute__((ext_vector_type(8)))  __bf16        v8bf;
typedef __attribute__((ext_vector_type(8)))  float         v8f;
typedef __attribute__((ext_vector_type(2)))  float         v2f;
typedef __attribute__((ext_vector_type(4)))  unsigned int  v4u;
typedef __attribute__((ext_vector_type(8)))  unsigned int  v8u;

constexpr int B_ = 8;
constexpr int K_ = 65536;
constexpr int C_ = 256;
constexpr int H_ = 128;
constexpr int T_ = 2048;
constexpr int D_ = 512;
constexpr int NBKT = 4096;
constexpr float LN_EPS = 1e-5f;

// LDS layout for the score kernel (dynamic shared memory blob):
//   w1t   : bf16 W1 transposed [H][264]                 -> 128*264*2 = 67,584 B
//   ft[2] : fp32 TDM-staged tiles [32][132] (TDM pads)  -> 2*16,896  = 33,792 B
//   bt[2] : bf16 transposed tiles [128][40]             -> 2*10,240  = 20,480 B
constexpr int W1T_PITCH = 264;          // bf16 elems per row (16B-aligned rows)
constexpr int FT_PITCH  = 132;          // dwords per row (TDM pad 128+4)
constexpr int BT_PITCH  = 40;           // bf16 elems per row (80 B rows)
constexpr unsigned OFS_FT0 = (unsigned)(H_ * W1T_PITCH * 2);            //  67584
constexpr unsigned OFS_FT1 = OFS_FT0 + (unsigned)(32 * FT_PITCH * 4);   //  84480
constexpr unsigned OFS_BT0 = OFS_FT1 + (unsigned)(32 * FT_PITCH * 4);   // 101376
constexpr unsigned OFS_BT1 = OFS_BT0 + (unsigned)(128 * BT_PITCH * 2);  // 111616
constexpr unsigned SCORE_LDS_BYTES = OFS_BT1 + (unsigned)(128 * BT_PITCH * 2);

// Issue one TDM: 128x32 fp32 tile (row stride K_) -> LDS, with TDM padding of
// 4 dwords after every 128 dwords (bank-conflict-free staged rows).
// D# per CDNA5 ISA (groups 0/1; groups 2/3 zero for 2D).
__device__ __forceinline__ void ggt_tdm_issue(const float* gsrc, unsigned lds_off)
{
    const unsigned long long ga = (unsigned long long)(const void*)gsrc;
    v4u g0;
    g0[0] = 1u;                                   // count=1, user mode
    g0[1] = lds_off;                              // lds_addr (bytes)
    g0[2] = (unsigned)ga;                         // global_addr[31:0]
    g0[3] = ((unsigned)(ga >> 32) & 0x01FFFFFFu)  // global_addr[56:32]
            | (2u << 30);                         // type=2 (image)
    v8u g1;
    g1[0] = 0x07920000u;  // data_size=4B | pad_en | interval=128dw | amount=4dw
    g1[1] = 0u;           // atomic_barrier_addr=0, tensor_dim0.lo16=0
    g1[2] = 0x01000001u;  // tensor_dim0.hi16=1 (64Ki), tensor_dim1.lo16=256
    g1[3] = 0x00800000u;  // tensor_dim1.hi16=0, tile_dim0=128
    g1[4] = 32u;          // tile_dim1=32, tile_dim2=0
    g1[5] = (unsigned)K_; // tensor_dim0_stride[31:0] = 65536
    g1[6] = 0u;           // stride0.hi16=0, stride1.lo16=0
    g1[7] = 0u;           // stride1.hi=0
    v4u g2 = {0u, 0u, 0u, 0u};
    v4u g3 = {0u, 0u, 0u, 0u};
    asm volatile("tensor_load_to_lds %0, %1, %2, %3"
                 :: "s"(g0), "s"(g1), "s"(g2), "s"(g3)
                 : "memory");
}

// ===========================================================================
// Kernel 1: graspability scores.
// grid = B*K/128 blocks, 256 threads (8 waves). Double-buffered pipeline per
// 32-channel C-tile:
//   wait tensorcnt -> barrier -> convert ft[p]->bt[p] (fp32->bf16 transpose)
//   -> issue TDM for next tile into ft[1-p] (overlaps with WMMA below)
//   -> barrier -> fragments as ds_load_b128 pairs, 8x v_wmma_f32_16x16x32_bf16
// Fused epilogue: ReLU -> dot(W2) -> sigmoid -> validity -> scores.
// ===========================================================================
__global__ __launch_bounds__(256)
void ggt_score_kernel(const float* __restrict__ xyz,
                      const float* __restrict__ features,
                      const float* __restrict__ W1,
                      const float* __restrict__ b1,
                      const float* __restrict__ W2,
                      const float* __restrict__ b2,
                      float* __restrict__ scores)
{
    extern __shared__ char smem[];
    __bf16* w1t = (__bf16*)smem;                  // [n][c] : w1t[n*264 + c]

    const int tid  = threadIdx.x;
    const int wave = tid >> 5;
    const int lane = tid & 31;
    const int half = lane >> 4;
    const int l16  = lane & 15;

    const int blocks_per_b = K_ / 128;
    const int b  = blockIdx.x / blocks_per_b;
    const int k0 = (blockIdx.x % blocks_per_b) * 128;

    // stage W1 (CxH fp32, row-major) -> LDS as transposed bf16 [H][C]
    for (int i = tid; i < C_ * H_; i += 256) {
        const int c = i >> 7, n = i & 127;
        w1t[n * W1T_PITCH + c] = (__bf16)W1[i];
    }

    const float* fb = features + (size_t)b * C_ * K_;
    // generic->LDS: low 32 bits of the flat address are the LDS byte offset
    const unsigned lds_base = (unsigned)(unsigned long long)(const void*)smem;

    v8f acc[8] = {};   // h tile: rows 0..15 (this wave), cols 0..127

    // prologue: DMA first tile
    if (wave == 0) ggt_tdm_issue(fb + k0, lds_base + OFS_FT0);

    for (int it = 0; it < C_ / 32; ++it) {
        const int c0 = it * 32;
        const int p  = it & 1;
        float*  ft = (float*)(smem + (p ? OFS_FT1 : OFS_FT0));
        __bf16* bt = (__bf16*)(smem + (p ? OFS_BT1 : OFS_BT0));

        if (wave == 0) __builtin_amdgcn_s_wait_tensorcnt(0);
        __syncthreads();   // ft[p] ready; iter it-1 convert reads all done

        // ---- convert + transpose: ft[c][k] fp32 -> bt[k][c] bf16 ----
        for (int e = tid; e < 32 * 128; e += 256) {
            const int c = e >> 7, k = e & 127;
            bt[k * BT_PITCH + c] = (__bf16)ft[c * FT_PITCH + k];
        }

        // overlap: DMA next tile into the other buffer while we do WMMA below
        if (wave == 0 && it + 1 < C_ / 32) {
            ggt_tdm_issue(fb + (size_t)(c0 + 32) * K_ + k0,
                          lds_base + (p ? OFS_FT0 : OFS_FT1));
        }
        __syncthreads();   // bt[p] ready (and w1t ready after first pass)

        // ---- A fragment: element i <-> kk = i + 16*half (contiguous c) ----
        const v8bf* ap = (const v8bf*)(bt + (wave * 16 + l16) * BT_PITCH + 16 * half);
        const v8bf alo = ap[0], ahi = ap[1];
        const v16bf a = __builtin_shufflevector(alo, ahi,
            0, 1, 2, 3, 4, 5, 6, 7, 8, 9, 10, 11, 12, 13, 14, 15);

        #pragma unroll
        for (int j = 0; j < 8; j++) {
            const int n = j * 16 + l16;
            const v8bf* bp = (const v8bf*)(w1t + n * W1T_PITCH + c0 + 16 * half);
            const v8bf blo = bp[0], bhi = bp[1];
            const v16bf bm = __builtin_shufflevector(blo, bhi,
                0, 1, 2, 3, 4, 5, 6, 7, 8, 9, 10, 11, 12, 13, 14, 15);
            acc[j] = __builtin_amdgcn_wmma_f32_16x16x32_bf16(
                false, a, false, bm, (short)0, acc[j], false, false);
        }
    }

    // ---- fused head: score[m] = sigmoid( sum_n relu(h+b1)*W2 + b2 ) ----
    // C layout: acc[j][r] is element (m = k0 + wave*16 + r + 8*half, n = 16j + l16)
    float s_r[8];
    #pragma unroll
    for (int r = 0; r < 8; r++) {
        float p = 0.f;
        #pragma unroll
        for (int j = 0; j < 8; j++) {
            const int n = j * 16 + l16;
            float hv = acc[j][r] + b1[n];
            hv = hv > 0.f ? hv : 0.f;
            p += hv * W2[n];
        }
        p += __shfl_xor(p, 1, 16);
        p += __shfl_xor(p, 2, 16);
        p += __shfl_xor(p, 4, 16);
        p += __shfl_xor(p, 8, 16);
        s_r[r] = p;
    }
    if (l16 == 0) {
        const float bb = b2[0];
        #pragma unroll
        for (int r = 0; r < 8; r++) {
            const int m = k0 + wave * 16 + r + 8 * half;
            float sc = 1.f / (1.f + __expf(-(s_r[r] + bb)));
            const float* xp = xyz + ((size_t)b * K_ + m) * 3;
            const float av = fabsf(xp[0]) + fabsf(xp[1]) + fabsf(xp[2]);
            scores[(size_t)b * K_ + m] = (av > 0.f) ? sc : -1.0f;
        }
    }
}

// ===========================================================================
// Kernels 2-5: bucket top-k (4096 bins over sigmoid range (0,1))
// ===========================================================================
__global__ void ggt_zero_kernel(unsigned int* __restrict__ p, int n)
{
    const int i = blockIdx.x * blockDim.x + threadIdx.x;
    if (i < n) p[i] = 0u;
}

__device__ __forceinline__ int ggt_bucket(float s)
{
    if (s < 0.f) return 0;                    // invalid sentinel -> lowest bin
    int bkt = (int)(s * (float)NBKT);
    return bkt > NBKT - 1 ? NBKT - 1 : bkt;
}

__global__ void ggt_hist_kernel(const float* __restrict__ scores,
                                unsigned int* __restrict__ hist)
{
    const int i = blockIdx.x * blockDim.x + threadIdx.x;   // over B*K
    const int b = i >> 16;
    atomicAdd(&hist[b * NBKT + ggt_bucket(scores[i])], 1u);
}

__global__ void ggt_scan_kernel(const unsigned int* __restrict__ hist,
                                unsigned int* __restrict__ rank_base)
{
    const int b = blockIdx.x;
    if (threadIdx.x == 0) {
        unsigned int run = 0;
        for (int bkt = NBKT - 1; bkt >= 0; bkt--) {        // descending rank
            rank_base[b * NBKT + bkt] = run;
            run += hist[b * NBKT + bkt];
        }
    }
}

__global__ void ggt_select_kernel(const float* __restrict__ scores,
                                  const unsigned int* __restrict__ rank_base,
                                  unsigned int* __restrict__ fill,
                                  int* __restrict__ sel_idx,
                                  int* __restrict__ sel_ok)
{
    const int i = blockIdx.x * blockDim.x + threadIdx.x;
    const int b = i >> 16;
    const int k = i & (K_ - 1);
    const float s = scores[i];
    const int bkt = ggt_bucket(s);
    const unsigned int pos  = atomicAdd(&fill[b * NBKT + bkt], 1u);
    const unsigned int slot = rank_base[b * NBKT + bkt] + pos;
    if (slot < (unsigned)T_) {
        sel_idx[b * T_ + slot] = k;
        sel_ok [b * T_ + slot] = (s >= 0.f) ? 1 : 0;
    }
}

// ===========================================================================
// Kernel 6: gather + token GEMM (fp32 WMMA 16x16x4) + LayerNorm + transpose
// grid = B*(T/16) blocks, 128 threads (4 waves). Each block: 16 tokens.
// Wave w covers output columns [w*128, w*128+128).
// ===========================================================================
__global__ __launch_bounds__(128)
void ggt_token_kernel(const float* __restrict__ xyz,
                      const float* __restrict__ features,
                      const float* __restrict__ Wa,
                      const float* __restrict__ ba,
                      const float* __restrict__ gamma,
                      const float* __restrict__ beta,
                      const int* __restrict__ sel_idx,
                      const int* __restrict__ sel_ok,
                      float* __restrict__ out_xyz,
                      float* __restrict__ out_feat)
{
    __shared__ float fs[16][C_ + 1];     // gathered features (bank-padded)
    __shared__ float os[16][D_ + 4];     // normalized tokens (pad=4: halves hit
                                         // disjoint bank ranges on store)
    __shared__ float redS[16][4], redQ[16][4];
    __shared__ float muS[16], rsS[16];
    __shared__ int   kidx[16], okf[16];

    const int tid  = threadIdx.x;        // 0..127
    const int wave = tid >> 5;
    const int lane = tid & 31;
    const int half = lane >> 4;
    const int l16  = lane & 15;

    const int b  = blockIdx.x / (T_ / 16);
    const int t0 = (blockIdx.x % (T_ / 16)) * 16;

    if (tid < 16) {
        kidx[tid] = sel_idx[b * T_ + t0 + tid];
        okf[tid]  = sel_ok [b * T_ + t0 + tid];
    }
    __syncthreads();

    // gather 16 tokens x 256 channels
    const float* fb = features + (size_t)b * C_ * K_;
    for (int e = tid; e < 16 * C_; e += 128) {
        const int t = e & 15, c = e >> 4;
        fs[t][c] = fb[(size_t)c * K_ + kidx[t]];
    }
    __syncthreads();

    const int nbase = wave * 128;
    v8f acc[8] = {};

    for (int c0 = 0; c0 < C_; c0 += 4) {
        // A: 16x4 fp32; lanes 0-15 hold K=0,1 ; lanes 16-31 hold K=2,3
        v2f a;
        a[0] = fs[l16][c0 + 2 * half];
        a[1] = fs[l16][c0 + 2 * half + 1];
        #pragma unroll
        for (int j = 0; j < 8; j++) {
            const int n = nbase + j * 16 + l16;
            v2f bm;
            bm[0] = Wa[(size_t)(c0 + 2 * half)     * D_ + n];
            bm[1] = Wa[(size_t)(c0 + 2 * half + 1) * D_ + n];
#if __has_builtin(__builtin_amdgcn_wmma_f32_16x16x4_f32)
            acc[j] = __builtin_amdgcn_wmma_f32_16x16x4_f32(
                false, a, false, bm, (short)0, acc[j], false, false);
#else
            // scalar fallback producing the same C-layout
            #pragma unroll
            for (int r = 0; r < 8; r++) {
                const int m = r + 8 * half;
                float s = acc[j][r];
                #pragma unroll
                for (int kk = 0; kk < 4; kk++)
                    s += fs[m][c0 + kk] * Wa[(size_t)(c0 + kk) * D_ + n];
                acc[j][r] = s;
            }
#endif
        }
    }

    // + bias (must precede LN statistics)
    #pragma unroll
    for (int j = 0; j < 8; j++) {
        const float bav = ba[nbase + j * 16 + l16];
        #pragma unroll
        for (int r = 0; r < 8; r++) acc[j][r] += bav;
    }

    // per-token partial sums over this wave's 128 columns, reduce over lanes
    #pragma unroll
    for (int r = 0; r < 8; r++) {
        float s = 0.f, q = 0.f;
        #pragma unroll
        for (int j = 0; j < 8; j++) { const float v = acc[j][r]; s += v; q += v * v; }
        s += __shfl_xor(s, 1, 16);  q += __shfl_xor(q, 1, 16);
        s += __shfl_xor(s, 2, 16);  q += __shfl_xor(q, 2, 16);
        s += __shfl_xor(s, 4, 16);  q += __shfl_xor(q, 4, 16);
        s += __shfl_xor(s, 8, 16);  q += __shfl_xor(q, 8, 16);
        if (l16 == 0) {
            const int t = r + 8 * half;
            redS[t][wave] = s;
            redQ[t][wave] = q;
        }
    }
    __syncthreads();
    if (tid < 16) {
        const float s  = redS[tid][0] + redS[tid][1] + redS[tid][2] + redS[tid][3];
        const float q  = redQ[tid][0] + redQ[tid][1] + redQ[tid][2] + redQ[tid][3];
        const float mu = s * (1.f / (float)D_);
        const float var = q * (1.f / (float)D_) - mu * mu;
        muS[tid] = mu;
        rsS[tid] = rsqrtf(var + LN_EPS);
    }
    __syncthreads();

    // normalize -> LDS tile (token-major), zero padded slots
    #pragma unroll
    for (int j = 0; j < 8; j++) {
        const int n = nbase + j * 16 + l16;
        const float g = gamma[n], be = beta[n];
        #pragma unroll
        for (int r = 0; r < 8; r++) {
            const int t = r + 8 * half;
            const float v = (acc[j][r] - muS[t]) * rsS[t] * g + be;
            os[t][n] = okf[t] ? v : 0.f;
        }
    }
    __syncthreads();

    // transpose-store: feat_out[b][n][t0..t0+15] contiguous 64B rows
    float* ofb = out_feat + (size_t)b * D_ * T_;
    for (int n = tid; n < D_; n += 128) {
        float tmp[16];
        #pragma unroll
        for (int t = 0; t < 16; t++) tmp[t] = os[t][n];
        float4* dst = (float4*)(ofb + (size_t)n * T_ + t0);
        const float4* src = (const float4*)tmp;
        dst[0] = src[0]; dst[1] = src[1]; dst[2] = src[2]; dst[3] = src[3];
    }

    // xyz_out
    if (tid < 16) {
        float x = 0.f, y = 0.f, z = 0.f;
        if (okf[tid]) {
            const float* xp = xyz + ((size_t)b * K_ + kidx[tid]) * 3;
            x = xp[0]; y = xp[1]; z = xp[2];
        }
        float* op = out_xyz + ((size_t)b * T_ + t0 + tid) * 3;
        op[0] = x; op[1] = y; op[2] = z;
    }
}

// ===========================================================================
extern "C" void kernel_launch(void* const* d_in, const int* in_sizes, int n_in,
                              void* d_out, int out_size, void* d_ws, size_t ws_size,
                              hipStream_t stream)
{
    const float* xyz      = (const float*)d_in[0];
    const float* features = (const float*)d_in[1];
    const float* W1       = (const float*)d_in[2];
    const float* b1       = (const float*)d_in[3];
    const float* W2       = (const float*)d_in[4];
    const float* b2       = (const float*)d_in[5];
    const float* Wa       = (const float*)d_in[6];
    const float* ba       = (const float*)d_in[7];
    const float* gamma    = (const float*)d_in[8];
    const float* beta     = (const float*)d_in[9];

    // workspace layout (hist and fill adjacent -> zeroed in one launch)
    char* ws = (char*)d_ws;
    float*        scores    = (float*)ws;        ws += (size_t)B_ * K_ * sizeof(float);
    unsigned int* hist      = (unsigned int*)ws; ws += (size_t)B_ * NBKT * sizeof(unsigned int);
    unsigned int* fill      = (unsigned int*)ws; ws += (size_t)B_ * NBKT * sizeof(unsigned int);
    unsigned int* rank_base = (unsigned int*)ws; ws += (size_t)B_ * NBKT * sizeof(unsigned int);
    int*          sel_idx   = (int*)ws;          ws += (size_t)B_ * T_ * sizeof(int);
    int*          sel_ok    = (int*)ws;          ws += (size_t)B_ * T_ * sizeof(int);
    if ((size_t)(ws - (char*)d_ws) > ws_size) return;   // insufficient scratch

    float* out_xyz  = (float*)d_out;
    float* out_feat = out_xyz + (size_t)B_ * T_ * 3;

    ggt_score_kernel<<<(B_ * K_) / 128, 256, SCORE_LDS_BYTES, stream>>>(
        xyz, features, W1, b1, W2, b2, scores);

    ggt_zero_kernel<<<(B_ * NBKT * 2 + 255) / 256, 256, 0, stream>>>(
        hist, B_ * NBKT * 2);

    ggt_hist_kernel<<<(B_ * K_) / 256, 256, 0, stream>>>(scores, hist);

    ggt_scan_kernel<<<B_, 32, 0, stream>>>(hist, rank_base);

    ggt_select_kernel<<<(B_ * K_) / 256, 256, 0, stream>>>(
        scores, rank_base, fill, sel_idx, sel_ok);

    ggt_token_kernel<<<B_ * (T_ / 16), 128, 0, stream>>>(
        xyz, features, Wa, ba, gamma, beta, sel_idx, sel_ok, out_xyz, out_feat);
}
